// RippleHead_18202071400561
// MI455X (gfx1250) — matrix operand order repaired and checked by reading, not compiled, not run on hardware
//
#include <hip/hip_runtime.h>
#include <hip/hip_bf16.h>
#include <stdint.h>

typedef __bf16 bf16_t;
typedef bf16_t v16bf __attribute__((ext_vector_type(16)));
typedef float  v8f   __attribute__((ext_vector_type(8)));

#define WMMA_BF16(A, B, C) \
  __builtin_amdgcn_wmma_f32_16x16x32_bf16(false, (A), false, (B), (short)0, (C), false, false)

// Problem constants (match reference setup_inputs)
constexpr int Bb   = 4;
constexpr int Tt   = 4096;
constexpr int Cdim = 1024;
constexpr int Hd   = 64;
constexpr int BT   = Bb * Tt;

// LDS strides in bf16 elements; all multiples of 8 so every 16B vector access is aligned.
constexpr int XS  = 40;   // x tile: 128 x 32 (row-major)
constexpr int WTS = 40;   // W tiles: stored TRANSPOSED as 64(n) x 32(k)
constexpr int KVS = 72;   // K tile: 128(key) x 64(h) row-major (144B rows)
constexpr int VTS = 136;  // V^T tile: 64(h) x 128(key) (272B rows)
constexpr int PSQ = 72;   // per-wave P scratch: 16 x 64 (+ pad)

union F16x16 { uint4 q[2]; v16bf v; };

__device__ __forceinline__ v8f vzero8() {
  v8f z = {0.f, 0.f, 0.f, 0.f, 0.f, 0.f, 0.f, 0.f};
  return z;
}

// Pack two f32 -> packed bf16 pair (low = a, high = b)
__device__ __forceinline__ unsigned pk2(float a, float b) {
  union { bf16_t h[2]; unsigned u; } x;
  x.h[0] = (bf16_t)a;
  x.h[1] = (bf16_t)b;
  return x.u;
}

// A-fragment (16x32 bf16) from row-major tile. Per ISA 7.12.2 the per-lane data is
// two contiguous 8-element runs: K = [8hi..8hi+7] and [16+8hi..16+8hi+7].
__device__ __forceinline__ v16bf load_A16(const bf16_t* t, int stride, int lane) {
  const int m  = lane & 15;
  const int hi = lane >> 4;
  const bf16_t* p = t + m * stride + 8 * hi;
  F16x16 u;
  u.q[0] = *reinterpret_cast<const uint4*>(p);
  u.q[1] = *reinterpret_cast<const uint4*>(p + 16);
  return u.v;
}

// B-fragment (32x16) where the SOURCE IS TRANSPOSED: Bt[n][kk] contiguous in kk.
// b[i] = Bt[n][K], K = [8hi..8hi+7] and [16+8hi..16+8hi+7] -> two b128 loads.
__device__ __forceinline__ v16bf load_Bt16(const bf16_t* t, int stride, int lane) {
  const int n  = lane & 15;
  const int hi = lane >> 4;
  const bf16_t* p = t + n * stride + 8 * hi;
  F16x16 u;
  u.q[0] = *reinterpret_cast<const uint4*>(p);
  u.q[1] = *reinterpret_cast<const uint4*>(p + 16);
  return u.v;
}

// Async 16B global->LDS DMA (CDNA5, ASYNCcnt-tracked).
__device__ __forceinline__ void async_b128(const bf16_t* gsrc, bf16_t* ldst) {
  const unsigned ldsa = (unsigned)(uintptr_t)ldst;
  const unsigned long long ga = (unsigned long long)(uintptr_t)gsrc;
  asm volatile("global_load_async_to_lds_b128 %0, %1, off"
               :: "v"(ldsa), "v"(ga) : "memory");
}

// -------------------- Kernel 1: fused QKV projection (bf16 out) --------------------
// grid = BT/128 blocks, 256 threads (8 waves). Wave w owns rows [r0+16w, r0+16w+16).
// q,k stored row-major [t][h]; v stored TRANSPOSED [b][h][t] (packed b128 stores).
__global__ void __launch_bounds__(256)
qkv_proj_kernel(const float* __restrict__ x,
                const float* __restrict__ Wq,
                const float* __restrict__ Wk,
                const float* __restrict__ Wv,
                bf16_t* __restrict__ qo,
                bf16_t* __restrict__ ko,
                bf16_t* __restrict__ vt) {
  __shared__ bf16_t xs[128 * XS];
  __shared__ bf16_t wt[3][64 * WTS];  // transposed: wt[m][n*WTS + k]

  const int tid  = threadIdx.x;
  const int lane = tid & 31;
  const int w    = tid >> 5;
  const int r0   = blockIdx.x * 128;
  const float* Wm[3] = {Wq, Wk, Wv};

  v8f accq[4], acck[4], accv[4];
#pragma unroll
  for (int nt = 0; nt < 4; ++nt) { accq[nt] = vzero8(); acck[nt] = vzero8(); accv[nt] = vzero8(); }

  for (int kc = 0; kc < Cdim; kc += 32) {
    __syncthreads();  // previous iteration's LDS readers done
    // stage x chunk: 128 rows x 32 cols (f32 -> packed bf16), 1024 float4s / 256 threads
#pragma unroll
    for (int e = 0; e < 4; ++e) {
      const int idx = tid + e * 256;
      const int row = idx >> 3, c4 = idx & 7;
      const float4 val = *reinterpret_cast<const float4*>(
          x + (size_t)(r0 + row) * Cdim + kc + c4 * 4);
      uint2 u;
      u.x = pk2(val.x, val.y);
      u.y = pk2(val.z, val.w);
      *reinterpret_cast<uint2*>(xs + row * XS + c4 * 4) = u;  // 8B-aligned
    }
    // stage W chunks TRANSPOSED: thread grabs 4 K-consecutive values at fixed n
    // (coalesced across threads along n), packs, stores one uint2 along k.
#pragma unroll
    for (int m = 0; m < 3; ++m) {
#pragma unroll
      for (int e = 0; e < 2; ++e) {
        const int g  = tid + e * 256;      // 512 groups of (n, 4k)
        const int n  = g & 63;
        const int k0 = (g >> 6) << 2;
        const float* src = Wm[m] + (size_t)(kc + k0) * Hd + n;
        const float f0 = src[0];
        const float f1 = src[Hd];
        const float f2 = src[2 * Hd];
        const float f3 = src[3 * Hd];
        uint2 u;
        u.x = pk2(f0, f1);
        u.y = pk2(f2, f3);
        *reinterpret_cast<uint2*>(wt[m] + n * WTS + k0) = u;  // 8B-aligned
      }
    }
    if (kc + 32 < Cdim) {  // hint next x chunk (global_prefetch_b8)
      __builtin_prefetch(x + (size_t)(r0 + (tid >> 1)) * Cdim + (kc + 32) + (tid & 1) * 16, 0, 0);
    }
    __syncthreads();

    const v16bf a = load_A16(xs + 16 * w * XS, XS, lane);
#pragma unroll
    for (int nt = 0; nt < 4; ++nt) {
      const v16bf bq = load_Bt16(wt[0] + nt * 16 * WTS, WTS, lane);
      accq[nt] = WMMA_BF16(a, bq, accq[nt]);
      const v16bf bk = load_Bt16(wt[1] + nt * 16 * WTS, WTS, lane);
      acck[nt] = WMMA_BF16(a, bk, acck[nt]);
      const v16bf bv = load_Bt16(wt[2] + nt * 16 * WTS, WTS, lane);
      accv[nt] = WMMA_BF16(a, bv, accv[nt]);
    }
  }

  const float qscale = 0.125f;  // 1/sqrt(64)
  const int n  = lane & 15;
  const int hi = lane >> 4;
  const int bidx = r0 / Tt;           // 128 | Tt: tile never crosses a batch
  const int tloc = r0 - bidx * Tt;    // t base within batch

  // q, k row-major (C/D layout: lane holds col n, rows r+8*hi)
#pragma unroll
  for (int nt = 0; nt < 4; ++nt) {
#pragma unroll
    for (int r = 0; r < 8; ++r) {
      const int row = r0 + 16 * w + r + 8 * hi;
      const size_t o = (size_t)row * Hd + nt * 16 + n;
      qo[o] = (bf16_t)(accq[nt][r] * qscale);
      ko[o] = (bf16_t)(acck[nt][r]);
    }
  }
  // v transposed [b][h][t]: rows r=0..7 are CONSECUTIVE t -> one packed b128 store.
#pragma unroll
  for (int nt = 0; nt < 4; ++nt) {
    union { uint4 qv; bf16_t h8[8]; } u;
#pragma unroll
    for (int r = 0; r < 8; ++r) u.h8[r] = (bf16_t)(accv[nt][r]);
    const int h = nt * 16 + n;
    bf16_t* dst = vt + (size_t)(bidx * Hd + h) * Tt + tloc + 16 * w + 8 * hi;
    *reinterpret_cast<uint4*>(dst) = u.qv;  // 16B-aligned
  }
}

// -------------------- Kernel 2: causal flash attention with ripple bias --------------------
// grid = Bb * (Tt/128) blocks, 256 threads (8 waves); wave w owns 16 query rows.
// Softmax processed in 64-key quads; K and V^T tiles staged with async b128 DMA.
__global__ void __launch_bounds__(256)
ripple_attn_kernel(const bf16_t* __restrict__ q,
                   const bf16_t* __restrict__ k,
                   const bf16_t* __restrict__ vt,
                   const float* __restrict__ decay,
                   float* __restrict__ out) {
  __shared__ bf16_t Ks[128 * KVS];     // row-major [key][h]
  __shared__ bf16_t VsT[64 * VTS];     // [h][key]
  __shared__ bf16_t Psc[8][16 * PSQ];  // per-wave 16x64 P scratch

  const int tid  = threadIdx.x;
  const int lane = tid & 31;
  const int w    = tid >> 5;
  const int n    = lane & 15;
  const int hi   = lane >> 4;

  const int tb    = blockIdx.x;
  const int b     = tb >> 5;      // Tt/128 = 32 tiles per batch
  const int tile  = tb & 31;
  const int tBase = tile * 128;   // query t base within batch
  const int rowT  = tBase + 16 * w;
  const float ad  = __builtin_fabsf(decay[0]);

  // Q A-fragments (16 rows x 64, two K=32 chunks); global bf16, 16B-aligned runs.
  const bf16_t* qrow = q + (size_t)(b * Tt + rowT) * Hd;
  v16bf aq[2];
#pragma unroll
  for (int c = 0; c < 2; ++c) {
    const bf16_t* p = qrow + (size_t)n * Hd + 32 * c + 8 * hi;
    F16x16 u;
    u.q[0] = *reinterpret_cast<const uint4*>(p);
    u.q[1] = *reinterpret_cast<const uint4*>(p + 16);
    aq[c] = u.v;
  }

  v8f O[4];
#pragma unroll
  for (int nt = 0; nt < 4; ++nt) O[nt] = vzero8();
  float mrun[8], lrun[8];
#pragma unroll
  for (int r = 0; r < 8; ++r) { mrun[r] = -__builtin_inff(); lrun[r] = 0.f; }

  const int nkb = tile + 1;  // causal truncation of key blocks
  for (int kb = 0; kb < nkb; ++kb) {
    __syncthreads();
    const bf16_t* kg  = k + (size_t)(b * Tt + kb * 128) * Hd;
    const bf16_t* vtg = vt + (size_t)(b * Hd) * Tt + kb * 128;  // + h*Tt + key

    // K tile: async DMA global->LDS, 16B per lane per op (1024 x 16B = 16KB).
#pragma unroll
    for (int e = 0; e < 4; ++e) {
      const int idx = tid + e * 256;
      const int row = idx >> 3, c8 = idx & 7;
      async_b128(kg + (size_t)row * Hd + c8 * 8, Ks + row * KVS + c8 * 8);
    }
    // V^T tile: async DMA too (rows are h, contiguous along key in global).
#pragma unroll
    for (int e = 0; e < 4; ++e) {
      const int idx = tid + e * 256;   // 64 h-rows x 16 uint4
      const int h = idx >> 4, c16 = idx & 15;
      async_b128(vtg + (size_t)h * Tt + c16 * 8, VsT + h * VTS + c16 * 8);
    }
    asm volatile("s_wait_asynccnt 0x0" ::: "memory");  // this wave's DMAs done
    __syncthreads();

    const bool diagBlk = (kb == tile);
#pragma unroll
    for (int qs = 0; qs < 2; ++qs) {         // 64-key quads
      if (diagBlk && (64 * qs > 16 * w + 15)) continue;  // wave-uniform fully-masked quad

      // ---- S = Q K^T for four 16-key subtiles, + bias/mask in f32 regs ----
      float S[4][8];
#pragma unroll
      for (int j = 0; j < 4; ++j) {
        const int colBase = qs * 64 + j * 16;
        if (diagBlk && (colBase > 16 * w + 15)) {  // wave-uniform dead subtile
#pragma unroll
          for (int r = 0; r < 8; ++r) S[j][r] = -__builtin_inff();
          continue;
        }
        v8f s = vzero8();
#pragma unroll
        for (int c = 0; c < 2; ++c) {
          // B[kk][n] = K[sub+n][c*32+kk]: contiguous in kk -> two ds_load_b128
          const v16bf bk = load_Bt16(Ks + colBase * KVS + c * 32, KVS, lane);
          s = WMMA_BF16(aq[c], bk, s);
        }
        const int kt = kb * 128 + colBase + n;  // key t (per-lane)
#pragma unroll
        for (int r = 0; r < 8; ++r) {
          const int qt = rowT + r + 8 * hi;
          float val = s[r];
          val = (kt <= qt) ? (val + (float)(kt - qt) * ad) : -__builtin_inff();
          S[j][r] = val;
        }
      }

      // ---- online softmax over the 64-key quad ----
#pragma unroll
      for (int r = 0; r < 8; ++r) {
        float mx = fmaxf(fmaxf(S[0][r], S[1][r]), fmaxf(S[2][r], S[3][r]));
        mx = fmaxf(mx, __shfl_xor(mx, 1));
        mx = fmaxf(mx, __shfl_xor(mx, 2));
        mx = fmaxf(mx, __shfl_xor(mx, 4));
        mx = fmaxf(mx, __shfl_xor(mx, 8));
        const float nm = fmaxf(mrun[r], mx);

        const float sc = __expf(mrun[r] - nm);
        lrun[r] *= sc;
#pragma unroll
        for (int nt = 0; nt < 4; ++nt) O[nt][r] *= sc;

        float s2 = 0.f;
#pragma unroll
        for (int j = 0; j < 4; ++j) {
          const float p = __expf(S[j][r] - nm);
          Psc[w][(r + 8 * hi) * PSQ + j * 16 + n] = (bf16_t)p;
          s2 += p;
        }
        s2 += __shfl_xor(s2, 1);
        s2 += __shfl_xor(s2, 2);
        s2 += __shfl_xor(s2, 4);
        s2 += __shfl_xor(s2, 8);
        lrun[r] += s2;
        mrun[r] = nm;
      }

      asm volatile("s_wait_dscnt 0x0" ::: "memory");  // P LDS RAW (same-wave transpose)

      // ---- O += P V over the quad (K = 64 in two 32-chunks) ----
#pragma unroll
      for (int c2 = 0; c2 < 2; ++c2) {
        const v16bf ap = load_A16(&Psc[w][c2 * 32], PSQ, lane);
#pragma unroll
        for (int nt = 0; nt < 4; ++nt) {
          // B[kk][n] = V[qs*64+c2*32+kk][nt*16+n] = VsT[nt*16+n][...]: contiguous
          const v16bf bv = load_Bt16(VsT + (nt * 16) * VTS + qs * 64 + c2 * 32, VTS, lane);
          O[nt] = WMMA_BF16(ap, bv, O[nt]);
        }
      }
    }
  }

  // ---- epilogue: normalize and store f32 ----
#pragma unroll
  for (int nt = 0; nt < 4; ++nt) {
#pragma unroll
    for (int r = 0; r < 8; ++r) {
      const int t = rowT + r + 8 * hi;
      out[(size_t)(b * Tt + t) * Hd + nt * 16 + n] = O[nt][r] / lrun[r];
    }
  }
}

// -------------------- host launch --------------------
extern "C" void kernel_launch(void* const* d_in, const int* in_sizes, int n_in,
                              void* d_out, int out_size, void* d_ws, size_t ws_size,
                              hipStream_t stream) {
  (void)in_sizes; (void)n_in; (void)out_size; (void)ws_size;
  const float* x     = (const float*)d_in[0];
  const float* Wq    = (const float*)d_in[1];
  const float* Wk    = (const float*)d_in[2];
  const float* Wv    = (const float*)d_in[3];
  const float* decay = (const float*)d_in[4];

  bf16_t* qws = (bf16_t*)d_ws;               // BT*Hd bf16 = 2 MB, row-major [t][h]
  bf16_t* kws = qws + (size_t)BT * Hd;       // +2 MB, row-major [t][h]
  bf16_t* vws = kws + (size_t)BT * Hd;       // +2 MB, TRANSPOSED [b][h][t]

  qkv_proj_kernel<<<BT / 128, 256, 0, stream>>>(x, Wq, Wk, Wv, qws, kws, vws);
  ripple_attn_kernel<<<Bb * (Tt / 128), 256, 0, stream>>>(qws, kws, vws, decay, (float*)d_out);
}
